// AttnBlock_13924283973966
// MI455X (gfx1250) — matrix operand order repaired and checked by reading, not compiled, not run on hardware
//
#include <hip/hip_runtime.h>
#include <hip/hip_bf16.h>

// ---------------------------------------------------------------------------
// Sinkhorn-bucket transformer forward for MI455X (gfx1250, wave32, WMMA).
// GEMMs: v_wmma_f32_16x16x32_f16, LDS double-buffered W tiles, pipelined A.
// W staging uses gfx1250 async global->LDS copies when available.
// ---------------------------------------------------------------------------

typedef __attribute__((ext_vector_type(16))) _Float16 v16h;
typedef __attribute__((ext_vector_type(8)))  _Float16 v8h;
typedef __attribute__((ext_vector_type(8)))  float    v8f;

#define NDIM   256
#define SEQ    8192
#define NB     128          // buckets per sequence (SEQ / BUCK)
#define BUCK   64
#define NH     8
#define DHD    32
#define NBATCH 4
#define MTOK   (NBATCH*SEQ) // 32768 tokens
#define FFD    1024
#define NLAYER 6
#define NBH    (NBATCH*NH)  // 32

#if defined(__AMDGCN__) && __has_builtin(__builtin_amdgcn_global_load_async_to_lds_b128) && __has_builtin(__builtin_amdgcn_s_wait_asynccnt)
#define USE_ASYNC_LDS 1
typedef int v4i_vs __attribute__((vector_size(16)));
typedef __attribute__((address_space(1))) v4i_vs* g_v4i_p;   // global int4*
typedef __attribute__((address_space(3))) v4i_vs* l_v4i_p;   // LDS int4*
#else
#define USE_ASYNC_LDS 0
#endif

// ---------------- device helpers ----------------

__device__ inline v8f v8f_zero() {
  v8f z = {0.f,0.f,0.f,0.f,0.f,0.f,0.f,0.f};
  return z;
}

__device__ inline v16h pack16(v8h lo, v8h hi) {
  v16h a;
#pragma unroll
  for (int e = 0; e < 8; ++e) { a[e] = lo[e]; a[e+8] = hi[e]; }
  return a;
}

// A operand fragment: 16(M) x 32(K) f16 tile, row-major source, ld elements.
// ISA layout: lane m=L&15, half=L>>4; elems 0..7 -> K=8*half+0..7,
// elems 8..15 -> K=16+8*half+0..7  (two contiguous 16B runs per lane).
__device__ inline v16h load_frag_a(const _Float16* base, int ld) {
  int lane = threadIdx.x & 31;
  int m = lane & 15, hf = lane >> 4;
  const _Float16* r = base + (size_t)m * ld + 8 * hf;
  v8h lo = *(const v8h*)(r);
  v8h hi = *(const v8h*)(r + 16);
  return pack16(lo, hi);
}

// B operand fragment: 32(K) x 16(N) f16 tile, row-major B[k][n], ld elements.
// ISA layout: lane n=L&15, half=L>>4; elems 0..7 -> K=8*half+0..7,
// elems 8..15 -> K=16+8*half+0..7 (strided gather down column n).
__device__ inline v16h load_frag_b(const _Float16* base, int ld) {
  int lane = threadIdx.x & 31;
  int n = lane & 15, hf = lane >> 4;
  const _Float16* r = base + n + (size_t)(8 * hf) * ld;
  v16h b;
#pragma unroll
  for (int e = 0; e < 8; ++e) { b[e] = r[(size_t)e * ld]; b[e+8] = r[(size_t)(16 + e) * ld]; }
  return b;
}

__device__ inline v8f wmma_f16(v16h a, v16h b, v8f c) {
  return __builtin_amdgcn_wmma_f32_16x16x32_f16(
      /*neg_a=*/false, a, /*neg_b=*/false, b,
      /*c_mod=*/(short)0, c, /*reuse_a=*/false, /*reuse_b=*/false);
}

__device__ inline float wave_sum(float v) {
#pragma unroll
  for (int o = 16; o; o >>= 1) v += __shfl_xor(v, o, 32);
  return v;
}

__device__ inline float gelu_f(float x) {
  return 0.5f * x * (1.0f + erff(x * 0.7071067811865476f));
}

// ---------------- kernels ----------------

__global__ void cvt_f32_to_f16(const float* __restrict__ in,
                               _Float16* __restrict__ out, int n) {
  int i = blockIdx.x * blockDim.x + threadIdx.x;
  if (i < n) out[i] = (_Float16)in[i];
}

// y[b,t,d] = x[b,d,t] + pe0[t/64][d] + pe1[t%64][d]
__global__ void embed_kernel(const float* __restrict__ x,
                             const float* __restrict__ pe0,
                             const float* __restrict__ pe1,
                             float* __restrict__ y) {
  int i  = blockIdx.x * blockDim.x + threadIdx.x;   // over MTOK*NDIM
  int d  = i & (NDIM - 1);
  int bt = i >> 8;
  int t  = bt & (SEQ - 1);
  int b  = bt >> 13;
  y[i] = x[((size_t)(b * NDIM + d)) * SEQ + t]
       + pe0[(t >> 6) * NDIM + d] + pe1[(t & 63) * NDIM + d];
}

// LayerNorm: one wave per row of 256, f16 output.
__global__ void ln_kernel(const float* __restrict__ x,
                          const float* __restrict__ g,
                          const float* __restrict__ bt,
                          _Float16* __restrict__ out) {
  int row  = blockIdx.x * (blockDim.x >> 5) + (threadIdx.x >> 5);
  int lane = threadIdx.x & 31;
  const float* xr = x + (size_t)row * NDIM;
  float v[8]; float s = 0.f;
#pragma unroll
  for (int e = 0; e < 8; ++e) { v[e] = xr[lane + 32*e]; s += v[e]; }
  float mean = wave_sum(s) * (1.f / NDIM);
  float q = 0.f;
#pragma unroll
  for (int e = 0; e < 8; ++e) { float d = v[e] - mean; q += d * d; }
  float inv = rsqrtf(wave_sum(q) * (1.f / NDIM) + 1e-5f);
  _Float16* orow = out + (size_t)row * NDIM;
#pragma unroll
  for (int e = 0; e < 8; ++e) {
    int d = lane + 32*e;
    orow[d] = (_Float16)((v[e] - mean) * inv * g[d] + bt[d]);
  }
}

// Final LayerNorm, writes transposed f32 output (b, d, t).
__global__ void ln_final_kernel(const float* __restrict__ x,
                                const float* __restrict__ g,
                                const float* __restrict__ bt,
                                float* __restrict__ out) {
  int row  = blockIdx.x * (blockDim.x >> 5) + (threadIdx.x >> 5);
  int lane = threadIdx.x & 31;
  const float* xr = x + (size_t)row * NDIM;
  float v[8]; float s = 0.f;
#pragma unroll
  for (int e = 0; e < 8; ++e) { v[e] = xr[lane + 32*e]; s += v[e]; }
  float mean = wave_sum(s) * (1.f / NDIM);
  float q = 0.f;
#pragma unroll
  for (int e = 0; e < 8; ++e) { float d = v[e] - mean; q += d * d; }
  float inv = rsqrtf(wave_sum(q) * (1.f / NDIM) + 1e-5f);
  int b = row >> 13, t = row & (SEQ - 1);
#pragma unroll
  for (int e = 0; e < 8; ++e) {
    int d = lane + 32*e;
    out[((size_t)(b * NDIM + d)) * SEQ + t] = (v[e] - mean) * inv * g[d] + bt[d];
  }
}

// Fused WMMA GEMM: C = act(A[MxK] @ W[KxN] + bias) + resid, f32 and/or f16 out.
// Block = 8 waves, tile 128(M) x 128(N). K consumed in 64-deep stages:
// W stage (64x128 f16) double-buffered in LDS (async copy when available),
// A fragments software-pipelined in registers. K, M, N exact multiples.
#define WT_LD 136   // 128 + 8 pad halfs; 272B row stride (16B aligned, conflict-light)

__global__ void gemm_kernel(const _Float16* __restrict__ A,
                            const _Float16* __restrict__ W,
                            const float* __restrict__ bias,
                            const float* __restrict__ resid,
                            float* __restrict__ Cf,
                            _Float16* __restrict__ Ch,
                            int M, int K, int N, int act) {
  __shared__ _Float16 wt[2][64 * WT_LD];
  int tid  = threadIdx.x;
  int wave = tid >> 5, lane = tid & 31;
  int m0 = blockIdx.x * 128 + wave * 16;
  int n0 = blockIdx.y * 128;

  v8f acc[8];
#pragma unroll
  for (int nt = 0; nt < 8; ++nt) acc[nt] = v8f_zero();

  int nst = K >> 6;   // number of 64-deep K stages

  // Cooperative W stage: 64 rows x 128 cols f16 = 1024 x 16B chunks, 4/thread.
  auto stage = [&](int s, int buf) {
    int k0 = s * 64;
#pragma unroll
    for (int c = 0; c < 4; ++c) {
      int idx = tid + 256 * c;
      int row = idx >> 4;            // 16 chunks per row
      int col = (idx & 15) * 8;
      const _Float16* src = W + (size_t)(k0 + row) * N + n0 + col;
      _Float16* dst = &wt[buf][row * WT_LD + col];
#if USE_ASYNC_LDS
      __builtin_amdgcn_global_load_async_to_lds_b128((g_v4i_p)src, (l_v4i_p)dst, 0, 0);
#else
      *(v8h*)dst = *(const v8h*)src;
#endif
    }
  };
  auto stage_wait = [&]() {
#if USE_ASYNC_LDS
    __builtin_amdgcn_s_wait_asynccnt(0);
#endif
  };

  // A fragments for one stage (two 16x32 k-steps), pipelined in registers.
  v8h alo0, ahi0, alo1, ahi1;
  auto load_a = [&](int s) {
    int m = lane & 15, hf = lane >> 4;
    const _Float16* r = A + (size_t)(m0 + m) * K + s * 64 + 8 * hf;
    alo0 = *(const v8h*)(r);
    ahi0 = *(const v8h*)(r + 16);
    alo1 = *(const v8h*)(r + 32);
    ahi1 = *(const v8h*)(r + 48);
  };

  stage(0, 0);
  load_a(0);
  stage_wait();
  __syncthreads();

  for (int s = 0; s < nst; ++s) {
    int buf = s & 1;
    v16h a0 = pack16(alo0, ahi0);
    v16h a1 = pack16(alo1, ahi1);
    if (s + 1 < nst) { stage(s + 1, buf ^ 1); load_a(s + 1); }
#pragma unroll
    for (int nt = 0; nt < 8; ++nt) {
      v16h b0 = load_frag_b(&wt[buf][nt * 16], WT_LD);
      acc[nt] = wmma_f16(a0, b0, acc[nt]);
    }
#pragma unroll
    for (int nt = 0; nt < 8; ++nt) {
      v16h b1 = load_frag_b(&wt[buf][32 * WT_LD + nt * 16], WT_LD);
      acc[nt] = wmma_f16(a1, b1, acc[nt]);
    }
    stage_wait();
    __syncthreads();
  }

  int hf = lane >> 4, nl = lane & 15;
#pragma unroll
  for (int nt = 0; nt < 8; ++nt) {
    int col = n0 + nt * 16 + nl;
    float bv = bias ? bias[col] : 0.f;
#pragma unroll
    for (int j = 0; j < 8; ++j) {
      int row = m0 + j + 8 * hf;
      float v = acc[nt][j] + bv;
      if (act) v = gelu_f(v);
      if (resid) v += resid[(size_t)row * N + col];
      if (Cf) Cf[(size_t)row * N + col] = v;
      if (Ch) Ch[(size_t)row * N + col] = (_Float16)v;
    }
  }
}

// Bucket summaries: sq/sk[bh, i, d] = mean over 64 tokens. One wave per (bh,i).
__global__ void summary_kernel(const _Float16* __restrict__ q,
                               const _Float16* __restrict__ kv,
                               float* __restrict__ sq, float* __restrict__ sk) {
  int pair = blockIdx.x * (blockDim.x >> 5) + (threadIdx.x >> 5);
  int lane = threadIdx.x & 31;           // d
  int i = pair & (NB - 1), bh = pair >> 7;
  int b = bh >> 3, h = bh & 7;
  const _Float16* qp = q  + ((size_t)(b * SEQ + i * BUCK)) * NDIM     + h * DHD + lane;
  const _Float16* kp = kv + ((size_t)(b * SEQ + i * BUCK)) * (2*NDIM) + h * DHD + lane;
  float aq = 0.f, ak = 0.f;
  for (int t = 0; t < BUCK; ++t) {
    aq += (float)qp[(size_t)t * NDIM];
    ak += (float)kp[(size_t)t * 2 * NDIM];
  }
  sq[(size_t)pair * DHD + lane] = aq * (1.f / BUCK);
  sk[(size_t)pair * DHD + lane] = ak * (1.f / BUCK);
}

// Top-1 routing: probs = softmax(R/0.75); top = max prob, idx = argmax (first hit).
__global__ void route_kernel(const float* __restrict__ sq,
                             const float* __restrict__ sk,
                             float* __restrict__ topv, int* __restrict__ topi) {
  int pair = blockIdx.x * (blockDim.x >> 5) + (threadIdx.x >> 5);
  int lane = threadIdx.x & 31;
  int bh = pair >> 7;
  const float* sqv = sq + (size_t)pair * DHD;
  float rloc[4];
  float best = -3.4e38f; int bidx = 0;
#pragma unroll
  for (int jj = 0; jj < 4; ++jj) {
    int j = lane + 32 * jj;
    const float* skv = sk + ((size_t)(bh * NB + j)) * DHD;
    float r = 0.f;
#pragma unroll
    for (int d = 0; d < DHD; ++d) r += sqv[d] * skv[d];
    r *= 0.1767766952966369f;            // dh^-0.5
    rloc[jj] = r;
    if (r > best) { best = r; bidx = j; }
  }
#pragma unroll
  for (int o = 16; o; o >>= 1) {
    float ov = __shfl_xor(best, o, 32);
    int   oi = __shfl_xor(bidx, o, 32);
    if (ov > best || (ov == best && oi < bidx)) { best = ov; bidx = oi; }
  }
  float s = 0.f;
#pragma unroll
  for (int jj = 0; jj < 4; ++jj) s += expf((rloc[jj] - best) * (1.f / 0.75f));
  s = wave_sum(s);
  if (lane == 0) { topv[pair] = 1.f / s; topi[pair] = bidx; }
}

// Per-bucket attention: dots = bq @ [top*K[idx] ; K[i]]^T, softmax, @ [top*V[idx]; V[i]].
// One 128-thread block per (bh, bucket). Dynamic LDS: Kt + Vs + dots + attnh.
#define ATTN_LDS_BYTES (8704 + 10240 + 32768 + 16384)  // 68096

__global__ void attn_kernel(const _Float16* __restrict__ q,
                            const _Float16* __restrict__ kv,
                            const float* __restrict__ topv,
                            const int* __restrict__ topi,
                            _Float16* __restrict__ o) {
  extern __shared__ char smem[];
  _Float16* Kt    = (_Float16*)smem;                 // [32][136]  K transposed (d, row)
  _Float16* Vs    = (_Float16*)(smem + 8704);        // [128][40]  V row-major
  float*    dots  = (float*)(smem + 18944);          // [64][128]
  _Float16* attnh = (_Float16*)(smem + 51712);       // [64][128]

  int i  = blockIdx.x;         // bucket
  int bh = blockIdx.y;
  int b = bh >> 3, h = bh & 7;
  int pair = bh * NB + i;
  float tv = topv[pair];
  int   j0 = topi[pair];
  int tid = threadIdx.x;

  // Stage K^T and V (rows 0..63 = routed bucket scaled by top, 64..127 = own).
  {
    int r = tid;
    int srcb = (r < 64) ? j0 : i;
    int tt   = srcb * BUCK + (r & 63);
    float sc = (r < 64) ? tv : 1.0f;
    const _Float16* kp = kv + ((size_t)(b * SEQ + tt)) * (2 * NDIM) + h * DHD;
    const _Float16* vp = kp + NDIM;
#pragma unroll
    for (int d0 = 0; d0 < DHD; d0 += 8) {
      v8h kk = *(const v8h*)(kp + d0);
      v8h vv = *(const v8h*)(vp + d0);
#pragma unroll
      for (int e = 0; e < 8; ++e) {
        Kt[(d0 + e) * 136 + r] = (_Float16)(sc * (float)kk[e]);
        Vs[r * 40 + d0 + e]    = (_Float16)(sc * (float)vv[e]);
      }
    }
  }
  __syncthreads();

  int wave = tid >> 5, lane = tid & 31;
  int hf = lane >> 4, nl = lane & 15;

  // dots = bq(64x32) @ K^T(32x128) * dh^-0.5 ; wave w handles rows w*16..+16.
  {
    const _Float16* abase =
        q + ((size_t)(b * SEQ + i * BUCK + wave * 16)) * NDIM + h * DHD;
    v16h a = load_frag_a(abase, NDIM);
#pragma unroll
    for (int nt = 0; nt < 8; ++nt) {
      v16h bfr = load_frag_b(Kt + nt * 16, 136);
      v8f c = wmma_f16(a, bfr, v8f_zero());
#pragma unroll
      for (int j = 0; j < 8; ++j)
        dots[(wave * 16 + j + 8 * hf) * 128 + nt * 16 + nl] =
            c[j] * 0.1767766952966369f;
    }
  }
  __syncthreads();

  // Row softmax (threads 0..63, one row each), emit f16 attn matrix.
  if (tid < 64) {
    float mx = -3.4e38f;
    for (int cc = 0; cc < 128; ++cc) mx = fmaxf(mx, dots[tid * 128 + cc]);
    float s = 0.f;
    for (int cc = 0; cc < 128; ++cc) {
      float e = expf(dots[tid * 128 + cc] - mx);
      dots[tid * 128 + cc] = e; s += e;
    }
    float inv = 1.f / s;
    for (int cc = 0; cc < 128; ++cc)
      attnh[tid * 128 + cc] = (_Float16)(dots[tid * 128 + cc] * inv);
  }
  __syncthreads();

  // out = attn(64x128) @ V(128x32)
  {
    v8f c0 = v8f_zero(), c1 = v8f_zero();
#pragma unroll
    for (int kc = 0; kc < 4; ++kc) {
      v16h a  = load_frag_a(attnh + (size_t)(wave * 16) * 128 + kc * 32, 128);
      v16h b0 = load_frag_b(Vs + (size_t)(kc * 32) * 40, 40);
      v16h b1 = load_frag_b(Vs + (size_t)(kc * 32) * 40 + 16, 40);
      c0 = wmma_f16(a, b0, c0);
      c1 = wmma_f16(a, b1, c1);
    }
    _Float16* op = o + ((size_t)(b * SEQ + i * BUCK + wave * 16)) * NDIM + h * DHD;
#pragma unroll
    for (int j = 0; j < 8; ++j) {
      int row = j + 8 * hf;
      op[(size_t)row * NDIM + nl]      = (_Float16)c0[j];
      op[(size_t)row * NDIM + 16 + nl] = (_Float16)c1[j];
    }
  }
}

// ---------------- host orchestration ----------------

extern "C" void kernel_launch(void* const* d_in, const int* in_sizes, int n_in,
                              void* d_out, int out_size, void* d_ws, size_t ws_size,
                              hipStream_t stream) {
  (void)in_sizes; (void)n_in; (void)out_size; (void)ws_size;
  const float* x    = (const float*)d_in[0];
  const float* pe0  = (const float*)d_in[1];
  const float* pe1  = (const float*)d_in[2];
  const float* ln1g = (const float*)d_in[3];
  const float* ln1b = (const float*)d_in[4];
  const float* Wq   = (const float*)d_in[5];
  const float* Wkv  = (const float*)d_in[6];
  const float* Wo   = (const float*)d_in[7];
  const float* bo   = (const float*)d_in[8];
  const float* ln2g = (const float*)d_in[9];
  const float* ln2b = (const float*)d_in[10];
  const float* W1   = (const float*)d_in[11];
  const float* b1   = (const float*)d_in[12];
  const float* W2   = (const float*)d_in[13];
  const float* b2   = (const float*)d_in[14];
  const float* gf   = (const float*)d_in[15];
  const float* bf   = (const float*)d_in[16];
  float* out = (float*)d_out;

  char* ws = (char*)d_ws;
  size_t off = 0;
  auto take = [&](size_t bytes) -> void* {
    void* p = ws + off;
    off = (off + bytes + 255) & ~(size_t)255;
    return p;
  };
  float*    y     = (float*)   take((size_t)MTOK * NDIM * 4);
  _Float16* h16   = (_Float16*)take((size_t)MTOK * NDIM * 2);
  _Float16* q16   = (_Float16*)take((size_t)MTOK * NDIM * 2);
  _Float16* kv16  = (_Float16*)take((size_t)MTOK * 2 * NDIM * 2);
  _Float16* o16   = (_Float16*)take((size_t)MTOK * NDIM * 2);
  _Float16* ff16  = (_Float16*)take((size_t)8192 * FFD * 2);
  _Float16* wq16  = (_Float16*)take((size_t)NLAYER * NDIM * NDIM * 2);
  _Float16* wkv16 = (_Float16*)take((size_t)NLAYER * NDIM * 2 * NDIM * 2);
  _Float16* wo16  = (_Float16*)take((size_t)NLAYER * NDIM * NDIM * 2);
  _Float16* w116  = (_Float16*)take((size_t)NLAYER * NDIM * FFD * 2);
  _Float16* w216  = (_Float16*)take((size_t)NLAYER * FFD * NDIM * 2);
  float*    sq    = (float*)   take((size_t)NBH * NB * DHD * 4);
  float*    sk    = (float*)   take((size_t)NBH * NB * DHD * 4);
  float*    topv  = (float*)   take((size_t)NBH * NB * 4);
  int*      topi  = (int*)     take((size_t)NBH * NB * 4);

  // Weight precision staging (f32 -> f16), done every call (deterministic).
  {
    int n;
    n = NLAYER * NDIM * NDIM;
    cvt_f32_to_f16<<<(n + 255) / 256, 256, 0, stream>>>(Wq, wq16, n);
    n = NLAYER * NDIM * 2 * NDIM;
    cvt_f32_to_f16<<<(n + 255) / 256, 256, 0, stream>>>(Wkv, wkv16, n);
    n = NLAYER * NDIM * NDIM;
    cvt_f32_to_f16<<<(n + 255) / 256, 256, 0, stream>>>(Wo, wo16, n);
    n = NLAYER * NDIM * FFD;
    cvt_f32_to_f16<<<(n + 255) / 256, 256, 0, stream>>>(W1, w116, n);
    n = NLAYER * FFD * NDIM;
    cvt_f32_to_f16<<<(n + 255) / 256, 256, 0, stream>>>(W2, w216, n);
  }

  embed_kernel<<<(size_t)MTOK * NDIM / 256, 256, 0, stream>>>(x, pe0, pe1, y);

  for (int l = 0; l < NLAYER; ++l) {
    ln_kernel<<<MTOK / 8, 256, 0, stream>>>(y, ln1g + l * NDIM, ln1b + l * NDIM, h16);

    gemm_kernel<<<dim3(MTOK / 128, NDIM / 128), 256, 0, stream>>>(
        h16, wq16 + (size_t)l * NDIM * NDIM, nullptr, nullptr, nullptr, q16,
        MTOK, NDIM, NDIM, 0);
    gemm_kernel<<<dim3(MTOK / 128, 2 * NDIM / 128), 256, 0, stream>>>(
        h16, wkv16 + (size_t)l * NDIM * 2 * NDIM, nullptr, nullptr, nullptr, kv16,
        MTOK, NDIM, 2 * NDIM, 0);

    summary_kernel<<<(NBH * NB) / 8, 256, 0, stream>>>(q16, kv16, sq, sk);
    route_kernel<<<(NBH * NB) / 8, 256, 0, stream>>>(sq, sk, topv, topi);
    attn_kernel<<<dim3(NB, NBH), 128, ATTN_LDS_BYTES, stream>>>(
        q16, kv16, topv, topi, o16);

    gemm_kernel<<<dim3(MTOK / 128, NDIM / 128), 256, 0, stream>>>(
        o16, wo16 + (size_t)l * NDIM * NDIM, bo + l * NDIM, y, y, nullptr,
        MTOK, NDIM, NDIM, 0);

    ln_kernel<<<MTOK / 8, 256, 0, stream>>>(y, ln2g + l * NDIM, ln2b + l * NDIM, h16);

    // Feed-forward in 4 row-chunks to keep the GELU intermediate at 16 MB.
    for (int c = 0; c < 4; ++c) {
      size_t ro = (size_t)c * 8192;
      gemm_kernel<<<dim3(8192 / 128, FFD / 128), 256, 0, stream>>>(
          h16 + ro * NDIM, w116 + (size_t)l * NDIM * FFD, b1 + l * FFD,
          nullptr, nullptr, ff16, 8192, NDIM, FFD, 1);
      gemm_kernel<<<dim3(8192 / 128, NDIM / 128), 256, 0, stream>>>(
          ff16, w216 + (size_t)l * FFD * NDIM, b2 + l * NDIM,
          y + ro * NDIM, y + ro * NDIM, nullptr, 8192, FFD, NDIM, 0);
    }
  }

  ln_final_kernel<<<MTOK / 8, 256, 0, stream>>>(y, gf, bf, out);
}
